// AugmentWithTrace_22058952032951
// MI455X (gfx1250) — matrix operand order repaired and checked by reading, not compiled, not run on hardware
//
#include <hip/hip_runtime.h>
#include <hip/hip_fp16.h>

typedef __attribute__((ext_vector_type(16))) _Float16 v16h;
typedef __attribute__((ext_vector_type(8)))  _Float16 v8h;
typedef __attribute__((ext_vector_type(8)))  float    v8f;

#define D_DIM 256
#define TOK_PER_BLK 8
#define ROWS 64               // TOK_PER_BLK * K
#define NEGV (-1e30f)

// 16-byte-block XOR swizzle: WMMA A-fragment loads have 16 lanes reading the
// same column block of 16 consecutive rows; XOR-ing the 8-half block index
// with (row & 15) spreads them across all 16 bank quads (conflict-free).
__device__ __forceinline__ int swz(int row, int c) {
    int blk = (c >> 3) ^ (row & 15);
    return row * D_DIM + (blk << 3) + (c & 7);
}

// One dense layer: ldsOut[64][256] = act(ldsIn[64][256] @ W + b), f16 in/out,
// f32 accumulate via v_wmma_f32_16x16x32_f16. Wt is the f16 TRANSPOSED weight
// (Wt[n][k] = W[k][n]) so each lane's B fragment is 32 contiguous bytes.
__device__ __forceinline__ void mlp_layer(
    const _Float16* ldsIn, _Float16* ldsOut,
    const _Float16* __restrict__ Wt, const float* __restrict__ bias,
    int lane, int wave, bool relu)
{
    const int ln = lane & 15;
    const int hi = lane >> 4;            // 0: lanes 0-15, 1: lanes 16-31
    const int n0 = wave * 32 + ln;       // first owned column
    const int n1 = n0 + 16;              // second owned column
    v8f acc[4][2] = {};

    #pragma unroll
    for (int ks = 0; ks < 8; ++ks) {
        const int kb = ks * 32;
        // B fragment: lanes 0-15 hold K=kb..kb+15 of column n, lanes 16-31 K=kb+16..kb+31
        v16h b0 = *(const v16h*)(Wt + (size_t)n0 * D_DIM + kb + hi * 16);
        v16h b1 = *(const v16h*)(Wt + (size_t)n1 * D_DIM + kb + hi * 16);
        #pragma unroll
        for (int mt = 0; mt < 4; ++mt) {
            const int row = mt * 16 + ln;
            // A fragment: lane<16 row M=ln holds K 0..15; lane>=16 K 16..31
            v8h alo = *(const v8h*)(ldsIn + swz(row, kb + hi * 16));
            v8h ahi = *(const v8h*)(ldsIn + swz(row, kb + hi * 16 + 8));
            v16h a;
            #pragma unroll
            for (int i = 0; i < 8; ++i) { a[i] = alo[i]; a[i + 8] = ahi[i]; }
            acc[mt][0] = __builtin_amdgcn_wmma_f32_16x16x32_f16(
                false, a, false, b0, (short)0, acc[mt][0], false, false);
            acc[mt][1] = __builtin_amdgcn_wmma_f32_16x16x32_f16(
                false, a, false, b1, (short)0, acc[mt][1], false, false);
        }
    }
    const float bv0 = bias[n0];
    const float bv1 = bias[n1];
    #pragma unroll
    for (int mt = 0; mt < 4; ++mt) {
        #pragma unroll
        for (int r = 0; r < 8; ++r) {
            const int m = mt * 16 + hi * 8 + r;    // D layout: VGPR r -> M=r (+8 for hi lanes)
            float v0 = acc[mt][0][r] + bv0;
            float v1 = acc[mt][1][r] + bv1;
            if (relu) { v0 = fmaxf(v0, 0.f); v1 = fmaxf(v1, 0.f); }
            ldsOut[swz(m, n0)] = (_Float16)v0;
            ldsOut[swz(m, n1)] = (_Float16)v1;
        }
    }
}

__global__ void __launch_bounds__(256) prep_weights_kernel(
    const float* __restrict__ W1, const float* __restrict__ W2,
    _Float16* __restrict__ W1t, _Float16* __restrict__ W2t)
{
    const int b = blockIdx.x;            // 0..511
    const int n = b & 255;
    const int k = threadIdx.x;
    if (b < 256) W1t[n * D_DIM + k] = (_Float16)W1[(size_t)k * D_DIM + n];
    else         W2t[n * D_DIM + k] = (_Float16)W2[(size_t)k * D_DIM + n];
}

__global__ void __launch_bounds__(256) attn_trace_kernel(
    const float* __restrict__ pool, const int* __restrict__ nidx,
    const unsigned char* __restrict__ nmask,
    const _Float16* __restrict__ W1t, const float* __restrict__ b1,
    const _Float16* __restrict__ W2t, const float* __restrict__ b2,
    const float* __restrict__ W3, const float* __restrict__ b3,
    float* __restrict__ out)
{
    __shared__ __align__(16) _Float16 bufA[ROWS * D_DIM];   // 32 KB
    __shared__ __align__(16) _Float16 bufB[ROWS * D_DIM];   // 32 KB  (total = 64 KB)
    const int tid  = threadIdx.x;
    const int lane = tid & 31, wave = tid >> 5;
    const int rowBase = blockIdx.x * ROWS;                  // flat (t*8+k) base

    // ---- Phase 0: gather g = pool[idx] into bufA as f16 (4 threads / row) ----
    {
        const int lrow = tid >> 2, part = tid & 3;
        const int idx = nidx[rowBase + lrow];
        const float* src = pool + (size_t)idx * D_DIM + part * 64;
        #pragma unroll
        for (int j = 0; j < 16; ++j) {
            const float4 v = *(const float4*)(src + j * 4);
            const int c = part * 64 + j * 4;
            _Float16* dst = bufA + swz(lrow, c);
            dst[0] = (_Float16)v.x; dst[1] = (_Float16)v.y;
            dst[2] = (_Float16)v.z; dst[3] = (_Float16)v.w;
        }
    }
    __syncthreads();

    // ---- Phase 1/2: two WMMA GEMM layers with ReLU (ping-pong LDS) ----
    mlp_layer(bufA, bufB, W1t, b1, lane, wave, true);   // h1 = relu(g @ W1 + b1)
    __syncthreads();
    mlp_layer(bufB, bufA, W2t, b2, lane, wave, true);   // h2 = relu(h1 @ W2 + b2)
    __syncthreads();

    // ---- Phase 3: scores + masked softmax (bufB is dead -> reuse for attn) ----
    float* attnw = (float*)bufB;
    if (tid < ROWS) {
        const int r = tid;
        float s = b3[0];
        #pragma unroll 4
        for (int blk = 0; blk < 32; ++blk) {
            const v8h h = *(const v8h*)(bufA + r * D_DIM + ((blk ^ (r & 15)) << 3));
            #pragma unroll
            for (int j = 0; j < 8; ++j) s += (float)h[j] * W3[blk * 8 + j];
        }
        const bool msk = nmask[rowBase + r] != 0;
        if (!msk) s = NEGV;
        // softmax over groups of 8 lanes (one token) via xor-shuffles
        float m = s;
        #pragma unroll
        for (int off = 1; off < 8; off <<= 1) m = fmaxf(m, __shfl_xor(m, off, 8));
        const float e = __expf(s - m);
        float sum = e;
        #pragma unroll
        for (int off = 1; off < 8; off <<= 1) sum += __shfl_xor(sum, off, 8);
        attnw[r] = msk ? (e / sum) : 0.f;   // all-masked token -> all zeros
    }
    __syncthreads();

    // ---- Phase 4: out[t][d] = sum_k attn * g  (re-gather f32 from L2-resident pool) ----
    {
        const int d = tid;                   // 0..255 = one output column per thread
        #pragma unroll 1
        for (int t = 0; t < TOK_PER_BLK; ++t) {
            float accv = 0.f;
            #pragma unroll
            for (int k = 0; k < 8; ++k) {
                const int r = t * 8 + k;
                const float w = attnw[r];
                const int idx = nidx[rowBase + r];
                accv = fmaf(w, pool[(size_t)idx * D_DIM + d], accv);
            }
            out[(size_t)(blockIdx.x * TOK_PER_BLK + t) * D_DIM + d] = accv;
        }
    }
}

extern "C" void kernel_launch(void* const* d_in, const int* in_sizes, int n_in,
                              void* d_out, int out_size, void* d_ws, size_t ws_size,
                              hipStream_t stream) {
    const float*         pool  = (const float*)d_in[0];
    const int*           nidx  = (const int*)d_in[1];
    const unsigned char* nmask = (const unsigned char*)d_in[2];   // jnp bool = 1 byte
    const float* W1 = (const float*)d_in[3];
    const float* b1 = (const float*)d_in[4];
    const float* W2 = (const float*)d_in[5];
    const float* b2 = (const float*)d_in[6];
    const float* W3 = (const float*)d_in[7];
    const float* b3 = (const float*)d_in[8];
    float* out = (float*)d_out;

    _Float16* W1t = (_Float16*)d_ws;                  // 128 KB
    _Float16* W2t = W1t + D_DIM * D_DIM;              // 128 KB

    prep_weights_kernel<<<512, 256, 0, stream>>>(W1, W2, W1t, W2t);

    const int T = in_sizes[1] / 8;                    // 32768 tokens
    attn_trace_kernel<<<T / TOK_PER_BLK, 256, 0, stream>>>(
        pool, nidx, nmask, W1t, b1, W2t, b2, W3, b3, out);
}
